// OnetoManyGRU_11828339933232
// MI455X (gfx1250) — compile-verified, matrix operand order checked by reading, not run on hardware
//
#include <hip/hip_runtime.h>

// ---------------------------------------------------------------------------
// One-to-many GRU decode + fused output projection, MI455X (gfx1250, wave32).
//
//   B=2048, E=256, K=128 steps, OUT=128.
//   s>=1:  g = h @ [W_ih; W_hh]^T  ([B,256] x [256,1536]) ; gates (r,z,n)
//          h' = (1-z)*n + z*h ;  out[:,s,:] = h' @ W_out^T + b_out
//   s==0:  x == 0, so gi == b_ih (the gi half of the GEMM is discarded).
//
// Residency plan (weights are step-invariant, so park them close):
//   * VGPRs: 16 fragments/wave (128 VGPRs) = gates 0,1 of col-group `wid`.
//   * LDS  : 192KB = gates 0..5, K 0..127 of col-groups 8..15  (+64KB W_out,
//            +32KB double-buffered hidden state) = 288KB of the 320KB WGP LDS.
//   * L2   : gates 2..5 of group `wid` + K-tail of group `wid+8` stream.
// Register-pressure control (rounds 2&3 spilled): streamed-fragment loops use
// `#pragma unroll 2` so the scheduler can only hoist ~2 iterations of loads;
// only the persistent-weight pass is fully unrolled (constant wp indices).
// ---------------------------------------------------------------------------

typedef __bf16 bf16_t;
typedef bf16_t v8bf  __attribute__((ext_vector_type(8)));
typedef bf16_t v16bf __attribute__((ext_vector_type(16)));
typedef float  v8f   __attribute__((ext_vector_type(8)));

#define E_DIM   256
#define BATCH   2048
#define KSTEPS  128
#define OUTD    128
#define BM      32           // batch rows per workgroup
#define THREADS 256          // 8 waves

__device__ __forceinline__ v16bf cat8(v8bf lo, v8bf hi) {
  return __builtin_shufflevector(lo, hi, 0,1,2,3,4,5,6,7,8,9,10,11,12,13,14,15);
}
// WMMA 16-bit operand fragment: lane l -> row/col (l&15), K = khalf..+7 and
// khalf+16..+23 (two contiguous 16B chunks).
__device__ __forceinline__ v16bf load_frag(const bf16_t* p) {
  v8bf lo = *(const v8bf*)p;
  v8bf hi = *(const v8bf*)(p + 16);
  return cat8(lo, hi);
}
__device__ __forceinline__ float fast_sigmoid(float x) {
  return __builtin_amdgcn_rcpf(1.f + __expf(-x));
}
__device__ __forceinline__ float fast_tanh(float x) {
  return 1.f - 2.f * __builtin_amdgcn_rcpf(__expf(2.f * x) + 1.f);
}
__device__ __forceinline__ v8f wmma_bf16(v16bf a, v16bf b, v8f c) {
  return __builtin_amdgcn_wmma_f32_16x16x32_bf16(false, a, false, b,
                                                 (short)0, c, false, false);
}
__device__ __forceinline__ v8f vzero() {
  return (v8f){0.f,0.f,0.f,0.f,0.f,0.f,0.f,0.f};
}

// ---------------------------------------------------------------------------
// Prep: Wc = [W_ih ; W_hh] as bf16, 1536 x 256 row-major, in workspace.
// ---------------------------------------------------------------------------
extern "C" __global__ void __launch_bounds__(256)
gru_prep(const float* __restrict__ Wih, const float* __restrict__ Whh,
         bf16_t* __restrict__ Wc) {
  int i = blockIdx.x * 256 + threadIdx.x;
  if (i < 3 * E_DIM * E_DIM) {
    Wc[i]                     = (bf16_t)Wih[i];
    Wc[3 * E_DIM * E_DIM + i] = (bf16_t)Whh[i];
  }
}

// Gate math for one 16-col group over both 16-row tiles; all in registers.
__device__ __forceinline__ void gate_update(
    const v8f (&acc)[2][6], float bir, float biz, float binn,
    float bhr, float bhz, float bhn, float gi_scale,
    int e, int rhalf, int cur, int nxt,
    bf16_t (&hb)[2][BM][E_DIM]) {
#pragma unroll
  for (int mt = 0; mt < 2; ++mt) {
#pragma unroll
    for (int i = 0; i < 8; ++i) {
      const int row = mt * 16 + rhalf + i;
      const float ir  = gi_scale * acc[mt][0][i] + bir;
      const float iz  = gi_scale * acc[mt][1][i] + biz;
      const float inn = gi_scale * acc[mt][2][i] + binn;
      const float hr  = acc[mt][3][i] + bhr;
      const float hz  = acc[mt][4][i] + bhz;
      const float hn  = acc[mt][5][i] + bhn;
      const float r   = fast_sigmoid(ir + hr);
      const float z   = fast_sigmoid(iz + hz);
      const float n   = fast_tanh(inn + r * hn);
      const float hp  = (float)hb[cur][row][e];
      hb[nxt][row][e] = (bf16_t)((1.f - z) * n + z * hp);
    }
  }
}

// ---------------------------------------------------------------------------
// Main persistent kernel: one workgroup = 32 batch rows for all 128 steps.
// Wave `wid` owns column groups {wid, wid+8} and OUT tile `wid`.
// ---------------------------------------------------------------------------
extern "C" __global__ void __launch_bounds__(256)
gru_decode(const float* __restrict__ c,
           const float* __restrict__ b_ih,
           const float* __restrict__ b_hh,
           const float* __restrict__ b_out,
           const float* __restrict__ WoutF,
           const bf16_t* __restrict__ Wc,
           float* __restrict__ out) {
  __shared__ bf16_t hbuf[2][BM][E_DIM];        //  32KB hidden, double-buffered
  __shared__ bf16_t wout_s[OUTD][E_DIM];       //  64KB projection weights
  __shared__ bf16_t wB_s[8][6][16][128];       // 192KB: groups 8..15, K 0..127

  const int b0    = blockIdx.x * BM;
  const int tid   = threadIdx.x;
  const int lane  = tid & 31;
  const int wid   = tid >> 5;                  // 0..7
  const int l15   = lane & 15;
  const int khalf = (lane >> 4) * 8;           // fragment K offset
  const int rhalf = (lane >> 4) * 8;           // C/D row offset

  // ---- One-time staging --------------------------------------------------
  for (int i = tid; i < OUTD * E_DIM; i += THREADS)
    ((bf16_t*)wout_s)[i] = (bf16_t)WoutF[i];

  // Gate weights for column groups 8..15, K 0..127, in [ng][gate][col][K]
  // fragment-friendly layout (per-col K contiguous).
  for (int i = tid; i < 8 * 6 * 16 * 128; i += THREADS) {
    const int t  = i >> 11;          // ng*6 + g
    const int g  = t % 6;
    const int ng = t / 6;
    const int j  = (i >> 7) & 15;
    const int k  = i & 127;
    ((bf16_t*)wB_s)[i] =
        Wc[(size_t)(g * E_DIM + 128 + ng * 16 + j) * E_DIM + k];
  }

  // Initial hidden state h0 = c into LDS buffer 0.
  for (int i = tid; i < BM * E_DIM; i += THREADS) {
    int r = i >> 8, col = i & (E_DIM - 1);
    hbuf[0][r][col] = (bf16_t)c[(size_t)(b0 + r) * E_DIM + col];
  }

  // VGPR-resident weight fragments: gates 0,1 of column-group `wid`,
  // all 8 K-iterations. 16 x v16bf = 128 VGPRs.
  v16bf wp[16];
#pragma unroll
  for (int kit = 0; kit < 8; ++kit)
#pragma unroll
    for (int g = 0; g < 2; ++g)
      wp[kit * 2 + g] =
          load_frag(Wc + (size_t)(g * E_DIM + wid * 16 + l15) * E_DIM +
                    kit * 32 + khalf);

  // Per-lane biases for both column groups (gates r,z,n).
  const int e0 = wid * 16 + l15;
  const int e1 = e0 + 128;
  const float bi0r = b_ih[e0],           bi1r = b_ih[e1];
  const float bi0z = b_ih[E_DIM + e0],   bi1z = b_ih[E_DIM + e1];
  const float bi0n = b_ih[2*E_DIM + e0], bi1n = b_ih[2*E_DIM + e1];
  const float bh0r = b_hh[e0],           bh1r = b_hh[e1];
  const float bh0z = b_hh[E_DIM + e0],   bh1z = b_hh[E_DIM + e1];
  const float bh0n = b_hh[2*E_DIM + e0], bh1n = b_hh[2*E_DIM + e1];
  const float bo   = b_out[wid * 16 + l15];

  __syncthreads();

  for (int s = 0; s < KSTEPS; ++s) {
    const int cur = s & 1, nxt = cur ^ 1;
    const float gi_scale = (s == 0) ? 0.f : 1.f;

    // ---- Phase A: column group `wid` --------------------------------------
    {
      v8f acc[2][6];
#pragma unroll
      for (int mt = 0; mt < 2; ++mt)
#pragma unroll
        for (int g = 0; g < 6; ++g) acc[mt][g] = vzero();

      // Pass 1: gates 0,1 from register-resident weights (full unroll so wp
      // indices stay compile-time constants).
#pragma unroll
      for (int kit = 0; kit < 8; ++kit) {
        const int kof = kit * 32 + khalf;
        v16bf a0 = load_frag(&hbuf[cur][l15][kof]);
        v16bf a1 = load_frag(&hbuf[cur][16 + l15][kof]);
        acc[0][0] = wmma_bf16(a0, wp[kit * 2 + 0], acc[0][0]);
        acc[1][0] = wmma_bf16(a1, wp[kit * 2 + 0], acc[1][0]);
        acc[0][1] = wmma_bf16(a0, wp[kit * 2 + 1], acc[0][1]);
        acc[1][1] = wmma_bf16(a1, wp[kit * 2 + 1], acc[1][1]);
      }
      // Pass 2: gates 2..5 streamed from L2 (bounded hoisting).
#pragma unroll 2
      for (int kit = 0; kit < 8; ++kit) {
        const int kof = kit * 32 + khalf;
        v16bf a0 = load_frag(&hbuf[cur][l15][kof]);
        v16bf a1 = load_frag(&hbuf[cur][16 + l15][kof]);
#pragma unroll
        for (int g = 2; g < 6; ++g) {
          v16bf bf = load_frag(
              Wc + (size_t)(g * E_DIM + wid * 16 + l15) * E_DIM + kof);
          acc[0][g] = wmma_bf16(a0, bf, acc[0][g]);
          acc[1][g] = wmma_bf16(a1, bf, acc[1][g]);
        }
      }
      gate_update(acc, bi0r, bi0z, bi0n, bh0r, bh0z, bh0n, gi_scale,
                  e0, rhalf, cur, nxt, hbuf);
    }

    // ---- Phase B: column group `wid+8`; K 0..127 from LDS, tail from L2 ---
    {
      v8f acc[2][6];
#pragma unroll
      for (int mt = 0; mt < 2; ++mt)
#pragma unroll
        for (int g = 0; g < 6; ++g) acc[mt][g] = vzero();

#pragma unroll 2
      for (int kit = 0; kit < 4; ++kit) {        // K 0..127 from LDS
        const int kof = kit * 32 + khalf;
        v16bf a0 = load_frag(&hbuf[cur][l15][kof]);
        v16bf a1 = load_frag(&hbuf[cur][16 + l15][kof]);
#pragma unroll
        for (int g = 0; g < 6; ++g) {
          v16bf bf = load_frag(&wB_s[wid][g][l15][kof]);
          acc[0][g] = wmma_bf16(a0, bf, acc[0][g]);
          acc[1][g] = wmma_bf16(a1, bf, acc[1][g]);
        }
      }
#pragma unroll 2
      for (int kit = 4; kit < 8; ++kit) {        // K 128..255 streamed
        const int kof = kit * 32 + khalf;
        v16bf a0 = load_frag(&hbuf[cur][l15][kof]);
        v16bf a1 = load_frag(&hbuf[cur][16 + l15][kof]);
#pragma unroll
        for (int g = 0; g < 6; ++g) {
          v16bf bf = load_frag(
              Wc + (size_t)(g * E_DIM + 128 + wid * 16 + l15) * E_DIM + kof);
          acc[0][g] = wmma_bf16(a0, bf, acc[0][g]);
          acc[1][g] = wmma_bf16(a1, bf, acc[1][g]);
        }
      }
      gate_update(acc, bi1r, bi1z, bi1n, bh1r, bh1z, bh1n, gi_scale,
                  e1, rhalf, cur, nxt, hbuf);
    }

    __syncthreads();   // h_next complete; old buffer free for next step

    // ---- Projection: out[:,s,:] = h_next @ Wout^T + b_out -----------------
#pragma unroll
    for (int mt = 0; mt < 2; ++mt) {
      v8f pacc = vzero();
#pragma unroll 2
      for (int kit = 0; kit < 8; ++kit) {
        const int kof = kit * 32 + khalf;
        v16bf afr = load_frag(&hbuf[nxt][mt * 16 + l15][kof]);
        v16bf bfr = load_frag(&wout_s[wid * 16 + l15][kof]);
        pacc = wmma_bf16(afr, bfr, pacc);
      }
#pragma unroll
      for (int i = 0; i < 8; ++i) {
        const size_t row = (size_t)(b0 + mt * 16 + rhalf + i);
        out[(row * KSTEPS + s) * OUTD + wid * 16 + l15] = pacc[i] + bo;
      }
    }
    // No second barrier: next step's LDS writes go to the buffer whose last
    // readers all completed before the barrier above.
  }
}

// ---------------------------------------------------------------------------
extern "C" void kernel_launch(void* const* d_in, const int* in_sizes, int n_in,
                              void* d_out, int out_size, void* d_ws, size_t ws_size,
                              hipStream_t stream) {
  const float* c    = (const float*)d_in[0];
  const float* Wih  = (const float*)d_in[1];
  const float* Whh  = (const float*)d_in[2];
  const float* bih  = (const float*)d_in[3];
  const float* bhh  = (const float*)d_in[4];
  const float* Wout = (const float*)d_in[5];
  const float* bout = (const float*)d_in[6];

  bf16_t* WcB = (bf16_t*)d_ws;                 // 1536*256 bf16 = 768KB
  float*  out = (float*)d_out;

  gru_prep<<<dim3((3 * E_DIM * E_DIM + 255) / 256), dim3(256), 0, stream>>>(
      Wih, Whh, WcB);
  gru_decode<<<dim3(BATCH / BM), dim3(THREADS), 0, stream>>>(
      c, bih, bhh, bout, Wout, WcB, out);
}
